// CausalCrossAttention_13013750907436
// MI455X (gfx1250) — matrix operand order repaired and checked by reading, not compiled, not run on hardware
//
#include <hip/hip_runtime.h>
#include <hip/hip_bf16.h>

// ---------------------------------------------------------------------------
// Causal cross-attention for MI455X (gfx1250, wave32).
//   B=4, S=2048, C=1024, NH=16, HD=64
// Pipeline:
//   [fp32->f16 convert: x, y, Wq, Wkv, Wo]  (one-shot, HBM-bound)
//   [Q proj GEMM]  : async-to-LDS staged, double-buffered, v_wmma f16
//   [KV proj GEMM] : same
//   [flash attention] : async Q/K staging, manual V transpose, online softmax
//   [O proj GEMM]  : fp32 output
// All matrix math on v_wmma_f32_16x16x32_f16; tile staging on
// global_load_async_to_lds_b128 (ASYNCcnt) per CDNA5 ISA.
// ---------------------------------------------------------------------------

typedef __attribute__((ext_vector_type(16))) _Float16 v16h;
typedef __attribute__((ext_vector_type(8)))  _Float16 v8h;
typedef __attribute__((ext_vector_type(4)))  _Float16 v4h;
typedef __attribute__((ext_vector_type(8)))  float    v8f;

#define ATTN_B  4
#define ATTN_S  2048
#define ATTN_C  1024
#define ATTN_NH 16
#define ATTN_HD 64

__device__ __forceinline__ v8f wmma_f16(v16h a, v16h b, v8f c) {
  return __builtin_amdgcn_wmma_f32_16x16x32_f16(
      false, a, false, b, (short)0, c, false, false);
}

__device__ __forceinline__ v8f zero8() {
  v8f z = {0.f, 0.f, 0.f, 0.f, 0.f, 0.f, 0.f, 0.f};
  return z;
}

// Async copy of 16 bytes global -> LDS (tracked by ASYNCcnt).
__device__ __forceinline__ void async_copy16(_Float16* lds_dst,
                                             const _Float16* gsrc) {
  unsigned lds = (unsigned)(uintptr_t)lds_dst;  // low 32 bits = LDS offset
  asm volatile("global_load_async_to_lds_b128 %0, %1, off"
               :: "v"(lds), "v"(gsrc)
               : "memory");
}
__device__ __forceinline__ void wait_async0() {
  asm volatile("s_wait_asynccnt 0x0" ::: "memory");
}

// A-fragment (16x32 MxK) from row-major LDS tile [row][k]:
// lane holds row M=lane%16; elems 0..7: K = kk+hh*8+e, elems 8..15: K = kk+16+hh*8+e.
__device__ __forceinline__ v16h ld_a(const _Float16* base, int row, int ld,
                                     int kk, int hh) {
  const _Float16* p = base + row * ld + kk + hh * 8;
  v8h lo = *(const v8h*)(p);
  v8h hi = *(const v8h*)(p + 16);
  v16h out;
#pragma unroll
  for (int i = 0; i < 8; ++i) { out[i] = lo[i]; out[i + 8] = hi[i]; }
  return out;
}

// B-fragment (32x16 KxN) from LDS tile stored [n][k] row-major:
// lane holds column N=lane%16; elems 0..7: K = kk+hh*8+e, elems 8..15: K = kk+16+hh*8+e.
__device__ __forceinline__ v16h ld_b(const _Float16* base, int row, int ld,
                                     int kk, int hh) {
  const _Float16* p = base + row * ld + kk + hh * 8;
  v8h lo = *(const v8h*)(p);
  v8h hi = *(const v8h*)(p + 16);
  v16h out;
#pragma unroll
  for (int i = 0; i < 8; ++i) { out[i] = lo[i]; out[i + 8] = hi[i]; }
  return out;
}

// ---------------------------------------------------------------------------
// fp32 -> f16 elementwise convert (n must be a multiple of 1024).
// ---------------------------------------------------------------------------
__global__ __launch_bounds__(256) void cvt_f32_f16_kernel(
    const float* __restrict__ in, _Float16* __restrict__ out) {
  size_t i = ((size_t)blockIdx.x * 256 + threadIdx.x) * 4;
  float4 f = *(const float4*)(in + i);
  v4h h;
  h[0] = (_Float16)f.x; h[1] = (_Float16)f.y;
  h[2] = (_Float16)f.z; h[3] = (_Float16)f.w;
  *(v4h*)(out + i) = h;
}

// ---------------------------------------------------------------------------
// GEMM: Out(MxN) = A(MxK f16) * W(NxK f16)^T + bias(N f32)
// 128x128 block tile, K-step 32, 256 threads = 8 waves (4x2), each wave 32x64.
// Tiles staged with global_load_async_to_lds_b128, double-buffered.
// ---------------------------------------------------------------------------
template <bool OUT_F16>
__global__ __launch_bounds__(256) void gemm_f16_kernel(
    const _Float16* __restrict__ A, const _Float16* __restrict__ W,
    const float* __restrict__ bias, void* __restrict__ Out,
    int M, int N, int K) {
  __shared__ _Float16 As[2][128 * 40];  // [m][k], ld=40
  __shared__ _Float16 Ws[2][128 * 40];  // [n][k], ld=40

  const int tid  = threadIdx.x;
  const int lane = tid & 31;
  const int w    = tid >> 5;
  const int wm   = w >> 1;     // 0..3
  const int wn   = w & 1;      // 0..1
  const int r    = lane & 15;
  const int hh   = lane >> 4;
  const int m0   = blockIdx.y * 128;
  const int n0   = blockIdx.x * 128;

  v8f acc[2][4];
#pragma unroll
  for (int i = 0; i < 2; ++i)
#pragma unroll
    for (int j = 0; j < 4; ++j) acc[i][j] = zero8();

  // stage one 128x32 A tile + 128x32 W tile into LDS buffer `buf` (async)
  auto stage = [&](int buf, int k0) {
#pragma unroll
    for (int i = 0; i < 2; ++i) {
      int c   = tid + i * 256;       // 512 b128 chunks per tile
      int row = c >> 2;
      int col = (c & 3) * 8;
      async_copy16(&As[buf][row * 40 + col],
                   A + (size_t)(m0 + row) * K + k0 + col);
      async_copy16(&Ws[buf][row * 40 + col],
                   W + (size_t)(n0 + row) * K + k0 + col);
    }
  };

  const int nk = K / 32;
  stage(0, 0);
  for (int kt = 0; kt < nk; ++kt) {
    wait_async0();        // this wave's outstanding copies done
    __syncthreads();      // all waves' copies visible
    if (kt + 1 < nk) stage((kt + 1) & 1, (kt + 1) * 32);
    const _Float16* a = As[kt & 1];
    const _Float16* ws = Ws[kt & 1];

    v16h a0 = ld_a(a, wm * 32 + 0  + r, 40, 0, hh);
    v16h a1 = ld_a(a, wm * 32 + 16 + r, 40, 0, hh);
#pragma unroll
    for (int tn = 0; tn < 4; ++tn) {
      v16h bf = ld_b(ws, wn * 64 + tn * 16 + r, 40, 0, hh);
      acc[0][tn] = wmma_f16(a0, bf, acc[0][tn]);
      acc[1][tn] = wmma_f16(a1, bf, acc[1][tn]);
    }
    // reads of this buffer complete before next iteration's barrier lets the
    // following stage() overwrite it (write to buf happens after that barrier)
  }

#pragma unroll
  for (int tm = 0; tm < 2; ++tm) {
#pragma unroll
    for (int tn = 0; tn < 4; ++tn) {
      int n    = n0 + wn * 64 + tn * 16 + r;
      float bv = bias[n];
#pragma unroll
      for (int j = 0; j < 8; ++j) {
        int m   = m0 + wm * 32 + tm * 16 + j + 8 * hh;
        float v = acc[tm][tn][j] + bv;
        if constexpr (OUT_F16)
          ((_Float16*)Out)[(size_t)m * N + n] = (_Float16)v;
        else
          ((float*)Out)[(size_t)m * N + n] = v;
      }
    }
  }
}

// ---------------------------------------------------------------------------
// Flash attention: one block per (b, h, 128-row q tile); 256 threads = 8 waves,
// wave w owns q rows [16w,16w+16). 64-key tiles, online softmax fp32.
// Q/K staged via async-to-LDS; V transposed manually; scale folded into S.
// ---------------------------------------------------------------------------
__global__ __launch_bounds__(256) void flash_attn_kernel(
    const _Float16* __restrict__ Q,    // [B*S, C]
    const _Float16* __restrict__ KV,   // [B*S, 2C]
    const unsigned char* __restrict__ mask,  // [B, S]
    _Float16* __restrict__ O) {        // [B*S, C]
  constexpr int S = ATTN_S, C = ATTN_C, HD = ATTN_HD;

  __shared__ _Float16 Qs[128 * 72];   // [qrow][d]
  __shared__ _Float16 Ks[64 * 72];    // [key][d]
  __shared__ _Float16 Vt[64 * 72];    // [d][key]
  __shared__ _Float16 Ps[128 * 72];   // [qrow][key]
  __shared__ float    mAdd[64];

  const int tid  = threadIdx.x;
  const int lane = tid & 31;
  const int w    = tid >> 5;
  const int r    = lane & 15;
  const int hh   = lane >> 4;
  const int q0   = blockIdx.x * 128;
  const int h    = blockIdx.y;
  const int b    = blockIdx.z;

  // ---- async-stage Q tile (pure copy; scale applied to S later) ----
#pragma unroll
  for (int i = 0; i < 2; ++i) {
    int c    = tid + i * 256;
    int row  = c >> 2;
    int part = (c & 3) * 8;
    async_copy16(&Qs[row * 72 + part],
                 Q + (size_t)(b * S + q0 + row) * C + h * HD + part);
  }

  float mi[8], li[8];
  v8f ofr[4];
#pragma unroll
  for (int j = 0; j < 8; ++j) { mi[j] = -1e30f; li[j] = 0.f; }
#pragma unroll
  for (int td = 0; td < 4; ++td) ofr[td] = zero8();

  const int   vrow  = tid >> 2;         // 0..63 key index for V/K staging
  const int   vpart = (tid & 3) * 8;    // d offset
  const float scale = 0.125f;           // 1/sqrt(64)

  const int kTiles = (q0 + 128) / 64;   // causal upper bound
  for (int t = 0; t < kTiles; ++t) {
    const int k0 = t * 64;
    const size_t gro =
        (size_t)(b * S + k0 + vrow) * (2 * C) + h * HD + vpart;
    // K tile: async pure copy into [key][d]
    async_copy16(&Ks[vrow * 72 + vpart], KV + gro);
    // V tile: load + transpose into [d][key]
    {
      v8h vf = *(const v8h*)(KV + gro + C);
#pragma unroll
      for (int e = 0; e < 8; ++e) Vt[(vpart + e) * 72 + vrow] = vf[e];
    }
    if (t + 1 < kTiles)  // prefetch next V tile (global_prefetch_b8)
      __builtin_prefetch(KV + gro + C + (size_t)64 * 2 * C, 0, 0);
    if (tid < 64)
      mAdd[tid] = mask[(size_t)b * S + k0 + tid] ? -1e30f : 0.0f;
    wait_async0();
    __syncthreads();

    // ---- S = Q * K^T (this wave: 16 q-rows x 64 keys) ----
    float sv[4][8];
    {
      v16h a0 = ld_a(Qs, w * 16 + r, 72, 0,  hh);
      v16h a1 = ld_a(Qs, w * 16 + r, 72, 32, hh);
#pragma unroll
      for (int tn = 0; tn < 4; ++tn) {
        v16h b0 = ld_b(Ks, tn * 16 + r, 72, 0,  hh);
        v16h b1 = ld_b(Ks, tn * 16 + r, 72, 32, hh);
        v8f s = zero8();
        s = wmma_f16(a0, b0, s);
        s = wmma_f16(a1, b1, s);
#pragma unroll
        for (int j = 0; j < 8; ++j) sv[tn][j] = s[j];
      }
    }

    // ---- scale + causal/padding mask ----
#pragma unroll
    for (int tn = 0; tn < 4; ++tn) {
      int   kpos = k0 + tn * 16 + r;
      float ma   = mAdd[tn * 16 + r];
#pragma unroll
      for (int j = 0; j < 8; ++j) {
        int   qpos = q0 + w * 16 + j + 8 * hh;
        float v    = fmaf(sv[tn][j], scale, ma);
        if (kpos > qpos) v = -1e30f;
        sv[tn][j] = v;
      }
    }

    // ---- online softmax (row stats over the 16-lane groups) ----
#pragma unroll
    for (int j = 0; j < 8; ++j) {
      float mx = fmaxf(fmaxf(sv[0][j], sv[1][j]), fmaxf(sv[2][j], sv[3][j]));
#pragma unroll
      for (int d = 1; d < 16; d <<= 1) mx = fmaxf(mx, __shfl_xor(mx, d, 32));
      float mnew = fmaxf(mi[j], mx);
      float sum = 0.f;
#pragma unroll
      for (int tn = 0; tn < 4; ++tn) {
        float p = __expf(sv[tn][j] - mnew);
        sv[tn][j] = p;
        sum += p;
      }
#pragma unroll
      for (int d = 1; d < 16; d <<= 1) sum += __shfl_xor(sum, d, 32);
      float alpha = __expf(mi[j] - mnew);
      li[j] = li[j] * alpha + sum;
      mi[j] = mnew;
#pragma unroll
      for (int td = 0; td < 4; ++td) ofr[td][j] *= alpha;
    }

    // ---- P: C-frag -> LDS -> A-frag ----
#pragma unroll
    for (int tn = 0; tn < 4; ++tn)
#pragma unroll
      for (int j = 0; j < 8; ++j)
        Ps[(w * 16 + j + 8 * hh) * 72 + tn * 16 + r] = (_Float16)sv[tn][j];
    __syncthreads();

    // ---- O += P * V ----
    {
      v16h a0 = ld_a(Ps, w * 16 + r, 72, 0,  hh);
      v16h a1 = ld_a(Ps, w * 16 + r, 72, 32, hh);
#pragma unroll
      for (int td = 0; td < 4; ++td) {
        v16h b0 = ld_b(Vt, td * 16 + r, 72, 0,  hh);
        v16h b1 = ld_b(Vt, td * 16 + r, 72, 32, hh);
        ofr[td] = wmma_f16(a0, b0, ofr[td]);
        ofr[td] = wmma_f16(a1, b1, ofr[td]);
      }
    }
    __syncthreads();  // before next iteration overwrites Ks/Vt/Ps
  }

  // ---- normalize, store f16 ----
#pragma unroll
  for (int j = 0; j < 8; ++j) {
    float inv  = 1.0f / li[j];
    size_t row = (size_t)(b * S + q0 + w * 16 + j + 8 * hh);
#pragma unroll
    for (int td = 0; td < 4; ++td)
      O[row * C + h * HD + td * 16 + r] = (_Float16)(ofr[td][j] * inv);
  }
}

// ---------------------------------------------------------------------------
extern "C" void kernel_launch(void* const* d_in, const int* in_sizes, int n_in,
                              void* d_out, int out_size, void* d_ws,
                              size_t ws_size, hipStream_t stream) {
  (void)in_sizes; (void)n_in; (void)out_size; (void)ws_size;
  const float*         x     = (const float*)d_in[0];
  const float*         y     = (const float*)d_in[1];
  const unsigned char* mask  = (const unsigned char*)d_in[2];
  const float*         Wq_w  = (const float*)d_in[3];
  const float*         Wq_b  = (const float*)d_in[4];
  const float*         Wkv_w = (const float*)d_in[5];
  const float*         Wkv_b = (const float*)d_in[6];
  const float*         Wo_w  = (const float*)d_in[7];
  const float*         Wo_b  = (const float*)d_in[8];
  float*               out   = (float*)d_out;

  const int    M  = ATTN_B * ATTN_S;   // 8192
  const int    C  = ATTN_C;            // 1024
  const size_t MC = (size_t)M * C;

  // workspace layout (f16 elements); xh region is reused as attention output
  _Float16* xh    = (_Float16*)d_ws;        // 16 MB (later: attn out)
  _Float16* yh    = xh + MC;                // 16 MB
  _Float16* Qbuf  = yh + MC;                // 16 MB
  _Float16* KVbuf = Qbuf + MC;              // 32 MB
  _Float16* Wqh   = KVbuf + 2 * MC;         // 2 MB
  _Float16* Wkvh  = Wqh + (size_t)C * C;    // 4 MB
  _Float16* Woh   = Wkvh + (size_t)2 * C * C;  // 2 MB
  _Float16* Obuf  = xh;                     // alias: x no longer needed

  dim3 blk(256);
  // one-shot fp32 -> f16 conversions (all sizes divisible by 1024)
  cvt_f32_f16_kernel<<<(unsigned)(MC / 1024), blk, 0, stream>>>(x, xh);
  cvt_f32_f16_kernel<<<(unsigned)(MC / 1024), blk, 0, stream>>>(y, yh);
  cvt_f32_f16_kernel<<<(unsigned)((size_t)C * C / 1024), blk, 0, stream>>>(Wq_w, Wqh);
  cvt_f32_f16_kernel<<<(unsigned)((size_t)2 * C * C / 1024), blk, 0, stream>>>(Wkv_w, Wkvh);
  cvt_f32_f16_kernel<<<(unsigned)((size_t)C * C / 1024), blk, 0, stream>>>(Wo_w, Woh);

  // Q = x @ Wq^T + b
  gemm_f16_kernel<true>
      <<<dim3(C / 128, M / 128), blk, 0, stream>>>(xh, Wqh, Wq_b, Qbuf,
                                                   M, C, C);
  // KV = y @ Wkv^T + b
  gemm_f16_kernel<true>
      <<<dim3(2 * C / 128, M / 128), blk, 0, stream>>>(yh, Wkvh, Wkv_b, KVbuf,
                                                       M, 2 * C, C);
  // attention (writes Obuf, which aliases xh -- x reads are done by now)
  flash_attn_kernel<<<dim3(ATTN_S / 128, ATTN_NH, ATTN_B), blk, 0, stream>>>(
      Qbuf, KVbuf, mask, Obuf);
  // out = attn @ Wo^T + b
  gemm_f16_kernel<false>
      <<<dim3(C / 128, M / 128), blk, 0, stream>>>(Obuf, Woh, Wo_b, out,
                                                   M, C, C);
}
